// MnistModel_66752381715115
// MI455X (gfx1250) — matrix operand order repaired and checked by reading, not compiled, not run on hardware
//
#include <hip/hip_runtime.h>
#include <hip/hip_bf16.h>

typedef __attribute__((ext_vector_type(16))) _Float16 v16h;
typedef __attribute__((ext_vector_type(8)))  float    v8f;

#define NPG 75
#define STRIDE 64   // fixed row stride for x/h buffers (max cout)

// ---------------------------------------------------------------------------
// rel[e] = (pos[src] - pos[tgt]) / RADIUS   (RADIUS = 2.0 -> *0.5)
// ---------------------------------------------------------------------------
__global__ void rel_kernel(const float* __restrict__ pos,
                           const int* __restrict__ src,
                           const int* __restrict__ tgt,
                           float* __restrict__ rel, int E) {
    int e = blockIdx.x * blockDim.x + threadIdx.x;
    if (e >= E) return;
    int s = src[e], t = tgt[e];
    rel[2 * e + 0] = (pos[2 * s + 0] - pos[2 * t + 0]) * 0.5f;
    rel[2 * e + 1] = (pos[2 * s + 1] - pos[2 * t + 1]) * 0.5f;
}

__global__ void zero_kernel(float* __restrict__ p, long n) {
    long i = (long)blockIdx.x * blockDim.x + threadIdx.x;
    if (i < n) p[i] = 0.0f;
}

// ---------------------------------------------------------------------------
// Pack W[cin,cout] (f32, row-major) -> Wt[coutp][cinp] (f16, transposed,
// zero-padded) and b[cout] -> bpad[coutp]. Puts B operands in this code's
// contiguous-k WMMA B order: 16 contiguous f16 per (tile,laneGroup) = 32B load.
// ---------------------------------------------------------------------------
__global__ void pack_w_kernel(const float* __restrict__ W, const float* __restrict__ b,
                              _Float16* __restrict__ Wt, float* __restrict__ bpad,
                              int cin, int cout, int cinp, int coutp) {
    int i = blockIdx.x * blockDim.x + threadIdx.x;
    int total = coutp * cinp;
    if (i < total) {
        int c = i / cinp, k = i % cinp;
        float v = (c < cout && k < cin) ? W[(long)k * cout + c] : 0.0f;
        Wt[(long)c * cinp + k] = (_Float16)v;
    }
    if (i < coutp) bpad[i] = (i < cout) ? b[i] : 0.0f;
}

// ---------------------------------------------------------------------------
// Node GEMM: h[Npad, COUTP] = (relu?)x[N, CIN] @ Wt^T + bpad
// One wave = one 16-row M-tile; grid covers Npad/16 tiles exactly, so all
// D-stores are unconditional (padding rows absorb the tail).
// A layout: element e -> k = k0 + 16*group + e
// B layout: 16 contiguous f16 at Wt[c*CINP + k0 + 16*group]
// ---------------------------------------------------------------------------
template <int CIN, int KSTEPS, int NTILES, bool RELU>
__global__ __launch_bounds__(256)
void node_gemm_kernel(const float* __restrict__ xin, int xstride,
                      const _Float16* __restrict__ Wt,
                      const float* __restrict__ bpad,
                      float* __restrict__ h, int Nn, int mtiles) {
    const int CINP = KSTEPS * 32;
    int wave = (blockIdx.x * blockDim.x + threadIdx.x) >> 5;
    int lane = threadIdx.x & 31;
    if (wave >= mtiles) return;          // wave-uniform: EXEC stays full
    __builtin_prefetch(Wt, 0, 1);        // global_prefetch path (weights hot)
    int row0 = wave * 16;
    int group = lane >> 4;
    int arow  = row0 + (lane & 15);
    if (arow >= Nn) arow = Nn - 1;       // clamp: input reads stay in-bounds
    int col   = lane & 15;

    // Build A tiles once, reuse across all N-tiles
    v16h a[KSTEPS];
#pragma unroll
    for (int ks = 0; ks < KSTEPS; ++ks) {
#pragma unroll
        for (int e = 0; e < 16; ++e) {
            int ka = ks * 32 + 16 * group + e;
            float av = xin[(long)arow * xstride + (ka < CIN ? ka : 0)];
            if (ka >= CIN) av = 0.0f;    // cndmask, not a branch
            if (RELU) av = __builtin_fmaxf(av, 0.0f);
            a[ks][e] = (_Float16)av;
        }
    }

#pragma unroll
    for (int nt = 0; nt < NTILES; ++nt) {
        int c = nt * 16 + col;
        v8f acc = {};
#pragma unroll
        for (int ks = 0; ks < KSTEPS; ++ks) {
            v16h bm = *(const v16h*)(Wt + (long)c * CINP + ks * 32 + 16 * group);
            acc = __builtin_amdgcn_wmma_f32_16x16x32_f16(
                false, a[ks], false, bm, (short)0, acc, false, false);
        }
        float bias = bpad[c];
#pragma unroll
        for (int r = 0; r < 8; ++r) {
            int rr = row0 + r + 8 * group;          // < Npad: unconditional
            h[(long)rr * STRIDE + c] = acc[r] + bias;
        }
    }
}

// ---------------------------------------------------------------------------
// Fused edge pipeline, 16 edges per wave:
//   t    = relu(rel @ F1 + fb1)   generated directly in WMMA-A layout (VALU)
//   filt = t @ F2t^T + fb2pad     (WMMA f16, K=64)
//   msg  = filt * h[src]          (gather + modulate)
//   xnext[tgt] += msg             (unconditional global_atomic_add_f32;
//                                  tail lanes contribute exactly 0.0)
// ---------------------------------------------------------------------------
template <int NTILES>
__global__ __launch_bounds__(256)
void edge_kernel(const float* __restrict__ rel,
                 const int* __restrict__ src, const int* __restrict__ tgt,
                 const float* __restrict__ F1, const float* __restrict__ fb1,
                 const _Float16* __restrict__ F2t,
                 const float* __restrict__ fb2pad,
                 const float* __restrict__ h, float* __restrict__ xnext,
                 int E, int etiles) {
    int wave = (blockIdx.x * blockDim.x + threadIdx.x) >> 5;
    int lane = threadIdx.x & 31;
    if (wave >= etiles) return;          // wave-uniform
    __builtin_prefetch(F2t, 0, 1);
    int e0 = wave * 16;
    int group = lane >> 4;
    int erow  = e0 + (lane & 15);
    if (erow >= E) erow = E - 1;         // clamp tail (read side)
    float rx = rel[2 * erow], ry = rel[2 * erow + 1];
    int col = lane & 15;

    // Filter hidden layer (K=2 MLP) directly in A layout; F1 is [2,64]
    v16h a[2];
#pragma unroll
    for (int ks = 0; ks < 2; ++ks) {
#pragma unroll
        for (int e = 0; e < 16; ++e) {
            int ka = ks * 32 + 16 * group + e;
            float tv = fmaf(rx, F1[ka], fmaf(ry, F1[64 + ka], fb1[ka]));
            a[ks][e] = (_Float16)__builtin_fmaxf(tv, 0.0f);
        }
    }

#pragma unroll
    for (int nt = 0; nt < NTILES; ++nt) {
        int c = nt * 16 + col;
        v8f acc = {};
#pragma unroll
        for (int ks = 0; ks < 2; ++ks) {
            v16h bm = *(const v16h*)(F2t + (long)c * 64 + ks * 32 + 16 * group);
            acc = __builtin_amdgcn_wmma_f32_16x16x32_f16(
                false, a[ks], false, bm, (short)0, acc, false, false);
        }
        float bias = fb2pad[c];
#pragma unroll
        for (int r = 0; r < 8; ++r) {
            int e = e0 + r + 8 * group;
            int ec = (e < E) ? e : (E - 1);         // clamp index (v_min)
            int s = src[ec], t = tgt[ec];
            float msg = (acc[r] + bias) * h[(long)s * STRIDE + c];
            if (e >= E) msg = 0.0f;                 // cndmask -> no-op add
            atomicAdd(&xnext[(long)t * STRIDE + c], msg);
        }
    }
}

// ---------------------------------------------------------------------------
// segment_max over each molecule's NPG nodes -> out[num_mols, 10]
// ---------------------------------------------------------------------------
__global__ void pool_kernel(const float* __restrict__ x, float* __restrict__ out,
                            int cout) {
    int mol = blockIdx.x;
    int c = threadIdx.x;
    if (c >= cout) return;
    float m = -3.402823466e38f;
    for (int n = 0; n < NPG; ++n) {
        float v = x[((long)mol * NPG + n) * STRIDE + c];
        m = v > m ? v : m;
    }
    out[mol * cout + c] = m;
}

extern "C" void kernel_launch(void* const* d_in, const int* in_sizes, int n_in,
                              void* d_out, int out_size, void* d_ws, size_t ws_size,
                              hipStream_t stream) {
    const int LAYERS = 8, HID = 64, OUT_DIM = 10;

    const float* features = (const float*)d_in[0];
    const float* pos      = (const float*)d_in[1];
    const int*   source   = (const int*)d_in[n_in - 4];
    const int*   target   = (const int*)d_in[n_in - 3];

    const int Nn   = in_sizes[0];          // IN_DIM == 1 -> flat count == N
    const int E    = in_sizes[n_in - 4];
    const int nmol = Nn / NPG;
    const int Npad = (Nn + 15) & ~15;      // padded rows absorb tail stores

    // ---- workspace layout (float units; keep 128B alignment) ----
    float* ws = (float*)d_ws;
    long o = 0;
    float* xA   = ws + o; o += (long)Npad * STRIDE;
    float* xB   = ws + o; o += (long)Npad * STRIDE;
    float* hbuf = ws + o; o += (long)Npad * STRIDE;
    float* rel  = ws + o; o += 2 * (((long)E + 15) & ~15L);
    float* bpad    = ws + o; o += 64;
    float* fb2pad  = ws + o; o += 64;
    _Float16* Wt   = (_Float16*)(ws + o); o += 64 * 64 / 2;  // 4096 halves
    _Float16* F2t  = (_Float16*)(ws + o); o += 64 * 64 / 2;

    rel_kernel<<<(E + 255) / 256, 256, 0, stream>>>(pos, source, target, rel, E);

    const float* xin = features;
    int xstride = 1;
    float* xcur = xA;

    const int mtiles = Npad / 16;
    const int etiles = (E + 15) / 16;
    const int gemm_blocks = (mtiles + 7) / 8;   // 8 waves per 256-thread block
    const int edge_blocks = (etiles + 7) / 8;

    for (int li = 0; li < LAYERS; ++li) {
        const int cout  = (li == LAYERS - 1) ? OUT_DIM : HID;
        const int coutp = (li == LAYERS - 1) ? 16 : 64;
        const int cin   = (li == 0) ? 1 : HID;
        const int cinp  = (li == 0) ? 32 : 64;

        const float* W   = (const float*)d_in[2 + 6 * li + 0];
        const float* b   = (const float*)d_in[2 + 6 * li + 1];
        const float* F1  = (const float*)d_in[2 + 6 * li + 2];
        const float* fb1 = (const float*)d_in[2 + 6 * li + 3];
        const float* F2  = (const float*)d_in[2 + 6 * li + 4];
        const float* fb2 = (const float*)d_in[2 + 6 * li + 5];

        // Pack weights for this layer (tiny)
        pack_w_kernel<<<(coutp * cinp + 255) / 256, 256, 0, stream>>>(
            W, b, Wt, bpad, cin, cout, cinp, coutp);
        pack_w_kernel<<<(coutp * 64 + 255) / 256, 256, 0, stream>>>(
            F2, fb2, F2t, fb2pad, HID, cout, 64, coutp);

        // Node GEMM (templated dispatch)
        if (li == 0) {
            node_gemm_kernel<1, 1, 4, false><<<gemm_blocks, 256, 0, stream>>>(
                xin, xstride, Wt, bpad, hbuf, Nn, mtiles);
        } else if (li == LAYERS - 1) {
            node_gemm_kernel<64, 2, 1, true><<<gemm_blocks, 256, 0, stream>>>(
                xin, xstride, Wt, bpad, hbuf, Nn, mtiles);
        } else {
            node_gemm_kernel<64, 2, 4, true><<<gemm_blocks, 256, 0, stream>>>(
                xin, xstride, Wt, bpad, hbuf, Nn, mtiles);
        }

        long nz = (long)Npad * STRIDE;
        zero_kernel<<<(int)((nz + 255) / 256), 256, 0, stream>>>(xcur, nz);

        if (li == LAYERS - 1) {
            edge_kernel<1><<<edge_blocks, 256, 0, stream>>>(
                rel, source, target, F1, fb1, F2t, fb2pad, hbuf, xcur, E, etiles);
        } else {
            edge_kernel<4><<<edge_blocks, 256, 0, stream>>>(
                rel, source, target, F1, fb1, F2t, fb2pad, hbuf, xcur, E, etiles);
        }

        xin = xcur; xstride = STRIDE;
        xcur = (xcur == xA) ? xB : xA;   // next layer writes the other buffer
    }

    // xin points at the last layer's output (raw segment sums, no relu)
    pool_kernel<<<nmol, 32, 0, stream>>>(xin, (float*)d_out, OUT_DIM);
}
